// BatchMoEDecoder_44547400794673
// MI455X (gfx1250) — matrix operand order repaired and checked by reading, not compiled, not run on hardware
//
#include <hip/hip_runtime.h>
#include <math.h>

// ---------------------------------------------------------------------------
// BatchMoEDecoder for MI455X (gfx1250, wave32, WMMA bf16)
//
// Sizes: B=1024 S=64 C=512 E=16 H=1024 D=256 V=100 K=2 NUM_NUM=32
// Strategy:
//   * one bandwidth-bound pre-pass converts all GEMM operands (weights,
//     codec, gathered pos/type) f32 -> bf16; GEMM hot loops then move raw
//     32-bit bf16 pairs global->LDS with no conversion VALU at all.
//   * gate MLP factorized: gate_in@gW1 = P1[b] + PT[s] (+ mask column)
//   * all GEMMs through one 128x128-tile WMMA kernel (8 waves, wave tile
//     32x64 = 2x4 v_wmma_f32_16x16x32_bf16 per 32-deep K-step), software
//     pipelined global->reg->LDS.
// Outputs (flat f32 elements in d_out, return order):
//   num_recon [B,32]        @ 0         (32768)
//   cat_recon [B,32,100]    @ 32768     (3276800)
//   gates     [B,S,2]       @ 3309568   (131072)
//   topi(int) [B,S,2]       @ 3440640   (131072, int bit-pattern)
//   mask(int) [B,S]         @ 3571712   (65536,  int bit-pattern)
// ---------------------------------------------------------------------------

typedef __attribute__((ext_vector_type(16))) __bf16 v16bf;
typedef __attribute__((ext_vector_type(8)))  float  v8f;

__device__ __forceinline__ unsigned short f32_to_bf16(float f) {
  unsigned int u = __float_as_uint(f);
  unsigned int r = u + 0x7FFFu + ((u >> 16) & 1u);   // round-to-nearest-even
  return (unsigned short)(r >> 16);
}
__device__ __forceinline__ float bf16_to_f32(unsigned short h) {
  return __uint_as_float(((unsigned int)h) << 16);
}
__device__ __forceinline__ unsigned int pack_bf16x2(float lo, float hi) {
  return (unsigned int)f32_to_bf16(lo) | ((unsigned int)f32_to_bf16(hi) << 16);
}

// store f32 result as f32 or bf16
template <typename T>
__device__ __forceinline__ void store_out(T* p, long long i, float v);
template <>
__device__ __forceinline__ void store_out<float>(float* p, long long i, float v) { p[i] = v; }
template <>
__device__ __forceinline__ void store_out<unsigned short>(unsigned short* p, long long i, float v) {
  p[i] = f32_to_bf16(v);
}

// ---------------------------------------------------------------------------
// f32 -> bf16 pre-pass (pair-packed, bandwidth-bound)
// ---------------------------------------------------------------------------
__global__ void cvt_f32_bf16_kernel(const float* __restrict__ in,
                                    unsigned short* __restrict__ outp,
                                    long long npairs) {
  long long i = (long long)blockIdx.x * blockDim.x + threadIdx.x;
  if (i >= npairs) return;
  float2 v = *(const float2*)(in + 2 * i);
  *(unsigned int*)(outp + 2 * i) = pack_bf16x2(v.x, v.y);
}

// ---------------------------------------------------------------------------
// Batched GEMM: D[z] = act(A[z][M,K](bf16) * B[z][K,N](bf16) + bias[z][N])
// OOB rows/cols are CLAMPED on load (always valid memory here) and the
// corresponding outputs discarded at store -> no divergence in staging.
// K must be a multiple of 32; ld's and tile coords are even so all 32-bit
// pair accesses are 4B aligned. ACT: 0 = identity, 1 = leaky_relu(0.01)
// ---------------------------------------------------------------------------
#define TILE_M 128
#define TILE_N 128
#define TILE_K 32
#define A_LDS 34   // 32 + 2 pad (aligned pair loads, conflict-free)
#define B_LDS 132  // 128 + 4 pad

template <typename TD, int ACT>
__global__ __launch_bounds__(256) void gemm_wmma_bf16(
    const unsigned short* __restrict__ Abase, long long strideA, int ldA,
    const unsigned short* __restrict__ Bbase, long long strideB, int ldB,
    const float* __restrict__ biasBase, long long strideBias,
    TD* __restrict__ Dbase, long long strideD, int ldD,
    int M, int N, int K) {
  __shared__ unsigned short As[TILE_M * A_LDS];
  __shared__ unsigned short Bs[TILE_K * B_LDS];

  const int z = blockIdx.z;
  const unsigned short* A = Abase + (long long)z * strideA;
  const unsigned short* Bw = Bbase + (long long)z * strideB;
  const float* bias = biasBase ? (biasBase + (long long)z * strideBias) : nullptr;
  TD* D = Dbase + (long long)z * strideD;

  const int tileM = blockIdx.y * TILE_M;
  const int tileN = blockIdx.x * TILE_N;
  const int tid  = threadIdx.x;
  const int lane = tid & 31;
  const int wid  = tid >> 5;
  const int wm   = wid & 3;   // 0..3 -> 32-row slab
  const int wn   = wid >> 2;  // 0..1 -> 64-col slab
  const int hi   = lane >> 4; // lane half (per ISA fragment layout)
  const int l15  = lane & 15;

  union F8 { v8f v; float f[8]; };
  union FragBF { v16bf v; unsigned int u[8]; };

  F8 acc[2][4];
#pragma unroll
  for (int mi = 0; mi < 2; ++mi)
#pragma unroll
    for (int ni = 0; ni < 4; ++ni)
#pragma unroll
      for (int r = 0; r < 8; ++r) acc[mi][ni].f[r] = 0.0f;

  // --- per-thread staging registers: 8 packed-pair words for A and for B ---
  unsigned int ra[8], rb[8];

  // A tile: 128x32 bf16 = 2048 words; word w -> row w>>4, k-pair (w&15)*2
  auto loadA = [&](int k0) {
#pragma unroll
    for (int j = 0; j < 8; ++j) {
      int w = tid + j * 256;
      int r = w >> 4, wc = w & 15;
      int gm = min(tileM + r, M - 1);            // clamp (valid memory)
      ra[j] = *(const unsigned int*)(A + (long long)gm * ldA + (k0 + 2 * wc));
    }
  };
  auto storeA = [&]() {
#pragma unroll
    for (int j = 0; j < 8; ++j) {
      int w = tid + j * 256;
      int r = w >> 4, wc = w & 15;
      *(unsigned int*)&As[r * A_LDS + 2 * wc] = ra[j];
    }
  };
  // B tile: 32x128 bf16 = 2048 words; word w -> row w>>6, n-pair (w&63)*2
  auto loadB = [&](int k0) {
#pragma unroll
    for (int j = 0; j < 8; ++j) {
      int w = tid + j * 256;
      int r = w >> 6, wc = w & 63;
      int cp = min(tileN + 2 * wc, N - 2);       // clamp pair base (aligned)
      rb[j] = *(const unsigned int*)(Bw + (long long)(k0 + r) * ldB + cp);
    }
  };
  auto storeB = [&]() {
#pragma unroll
    for (int j = 0; j < 8; ++j) {
      int w = tid + j * 256;
      int r = w >> 6, wc = w & 63;
      *(unsigned int*)&Bs[r * B_LDS + 2 * wc] = rb[j];
    }
  };

  loadA(0);
  loadB(0);

  for (int k0 = 0; k0 < K; k0 += TILE_K) {
    storeA();
    storeB();
    __syncthreads();

    // issue next tile's global loads; they complete while WMMAs run
    if (k0 + TILE_K < K) {
      loadA(k0 + TILE_K);
      loadB(k0 + TILE_K);
    }

    // ---- build fragments per documented VGPR layouts ----
    FragBF fa[2], fb[4];
#pragma unroll
    for (int mi = 0; mi < 2; ++mi) {
      int ms = wm * 32 + mi * 16 + l15;  // A: lane 0..15 carries row M=lane
#pragma unroll
      for (int v2 = 0; v2 < 8; ++v2) {
        // VGPR v holds K-pair {0,2,4,6,16,18,20,22}[v] (+8 for lanes 16..31)
        int kb = ((v2 < 4) ? 2 * v2 : 16 + 2 * (v2 - 4)) + 8 * hi;
        fa[mi].u[v2] = *(const unsigned int*)&As[ms * A_LDS + kb];
      }
    }
#pragma unroll
    for (int ni = 0; ni < 4; ++ni) {
      int ns = wn * 64 + ni * 16;  // B: lane carries N = lane&15, VGPR v -> K pair same as A
#pragma unroll
      for (int v2 = 0; v2 < 8; ++v2) {
        int kb = ((v2 < 4) ? 2 * v2 : 16 + 2 * (v2 - 4)) + 8 * hi;
        fb[ni].u[v2] = (unsigned int)Bs[kb * B_LDS + ns + l15] |
                       ((unsigned int)Bs[(kb + 1) * B_LDS + ns + l15] << 16);
      }
    }

#pragma unroll
    for (int mi = 0; mi < 2; ++mi)
#pragma unroll
      for (int ni = 0; ni < 4; ++ni)
        acc[mi][ni].v = __builtin_amdgcn_wmma_f32_16x16x32_bf16(
            false, fa[mi].v, false, fb[ni].v, (short)0, acc[mi][ni].v, false, false);
    __syncthreads();
  }

  // ---- epilogue: bias + activation + store (C layout: VGPR r -> M, lane -> N) ----
#pragma unroll
  for (int mi = 0; mi < 2; ++mi) {
#pragma unroll
    for (int ni = 0; ni < 4; ++ni) {
      int ncol = tileN + wn * 64 + ni * 16 + l15;
      float bv = 0.0f;
      if (bias && ncol < N) bv = bias[ncol];
#pragma unroll
      for (int r = 0; r < 8; ++r) {
        int mrow = tileM + wm * 32 + mi * 16 + r + 8 * hi;
        if (mrow < M && ncol < N) {
          float x = acc[mi][ni].f[r] + bv;
          if (ACT == 1) x = (x > 0.0f) ? x : 0.01f * x;
          store_out(D, (long long)mrow * ldD + ncol, x);
        }
      }
    }
  }
}

// ---------------------------------------------------------------------------
// Apt[s, 0:512] = pos_emb[s]; Apt[s, 512:1024] = type_emb[s>=32]   (bf16 out)
// ---------------------------------------------------------------------------
__global__ void build_apt_kernel(const float* __restrict__ pos_emb,
                                 const float* __restrict__ type_emb,
                                 unsigned short* __restrict__ apt) {
  int idx = blockIdx.x * 256 + threadIdx.x;  // 64*1024
  int s = idx >> 10, c = idx & 1023;
  float v = (c < 512) ? pos_emb[s * 512 + c]
                      : type_emb[((s >= 32) ? 1 : 0) * 512 + (c - 512)];
  apt[idx] = f32_to_bf16(v);
}

// ---------------------------------------------------------------------------
// Gate: h = gelu(P1[b] + PT[s] + mask*gW1[3C] + gb1); logits = (h@gW2+gb2)*mask
// top-2 + softmax -> gates/topi. One wave per token, 8 tokens per block.
// ---------------------------------------------------------------------------
__global__ __launch_bounds__(256) void gate_topk_kernel(
    const float* __restrict__ P1, const float* __restrict__ PT,
    const float* __restrict__ gW1, const float* __restrict__ gb1,
    const float* __restrict__ gW2, const float* __restrict__ gb2,
    const int* __restrict__ mask_pos,
    float* __restrict__ gates_out, int* __restrict__ topi_out) {
  __shared__ float hbuf[8][256];
  __shared__ float lbuf[8][16];
  const int wave = threadIdx.x >> 5;
  const int lane = threadIdx.x & 31;
  const int token = blockIdx.x * 8 + wave;  // < 65536
  const int b = token >> 6, s = token & 63;
  const float maskf = (float)mask_pos[token];
  const float* wmask = gW1 + 1536 * 256;

#pragma unroll
  for (int t = 0; t < 8; ++t) {
    int j = lane * 8 + t;
    float x = P1[b * 256 + j] + PT[s * 256 + j] + maskf * wmask[j] + gb1[j];
    // exact GeLU (approximate=False)
    hbuf[wave][j] = 0.5f * x * (1.0f + erff(x * 0.70710678118654752f));
  }
  __syncthreads();
  if (lane < 16) {
    float acc = gb2[lane];
    for (int k2 = 0; k2 < 256; ++k2) acc += hbuf[wave][k2] * gW2[k2 * 16 + lane];
    lbuf[wave][lane] = acc * maskf;
  }
  __syncthreads();
  if (lane == 0) {
    float v0 = -3.0e38f, v1 = -3.0e38f;
    int i0 = 0, i1 = 0;
    for (int e = 0; e < 16; ++e) {
      float v = lbuf[wave][e];
      if (v > v0) { v1 = v0; i1 = i0; v0 = v; i0 = e; }
      else if (v > v1) { v1 = v; i1 = e; }
    }
    float e1 = __expf(v1 - v0);            // softmax over {v0, v1}
    float inv = 1.0f / (1.0f + e1);
    gates_out[token * 2 + 0] = inv;
    gates_out[token * 2 + 1] = e1 * inv;
    topi_out[token * 2 + 0] = i0;
    topi_out[token * 2 + 1] = i1;
  }
}

// ---------------------------------------------------------------------------
// combined[b,s,d] = g0*eout[i0,b,d] + g1*eout[i1,b,d]   (bf16 out)
// ---------------------------------------------------------------------------
__global__ void combine_kernel(const float* __restrict__ eout,
                               const float* __restrict__ gates,
                               const int* __restrict__ topi,
                               unsigned short* __restrict__ combined) {
  const int token = blockIdx.x;   // B*S
  const int d = threadIdx.x;      // D=256
  const int b = token >> 6;
  float g0 = gates[token * 2 + 0], g1 = gates[token * 2 + 1];
  int i0 = topi[token * 2 + 0], i1 = topi[token * 2 + 1];
  float v = g0 * eout[((long long)i0 * 1024 + b) * 256 + d] +
            g1 * eout[((long long)i1 * 1024 + b) * 256 + d];
  combined[(long long)token * 256 + d] = f32_to_bf16(v);
}

// ---------------------------------------------------------------------------
// num_recon[b,n] = dot(combined[b,n,:], nW[n,:]) + nb[n]   (n < 32)
// ---------------------------------------------------------------------------
__global__ void num_head_kernel(const unsigned short* __restrict__ combined,
                                const float* __restrict__ nW,
                                const float* __restrict__ nb,
                                float* __restrict__ out) {
  int idx = blockIdx.x * blockDim.x + threadIdx.x;  // B*32
  if (idx >= 1024 * 32) return;
  int b = idx >> 5, n = idx & 31;
  const unsigned short* c = combined + ((long long)(b * 64 + n)) * 256;
  const float* w = nW + n * 256;
  float acc = nb[n];
  for (int d = 0; d < 256; ++d) acc += bf16_to_f32(c[d]) * w[d];
  out[idx] = acc;
}

// ---------------------------------------------------------------------------
extern "C" void kernel_launch(void* const* d_in, const int* in_sizes, int n_in,
                              void* d_out, int out_size, void* d_ws, size_t ws_size,
                              hipStream_t stream) {
  (void)in_sizes; (void)n_in; (void)out_size; (void)ws_size;
  const float* codec   = (const float*)d_in[0];
  const int*   maskp   = (const int*)d_in[1];
  const float* pos_emb = (const float*)d_in[2];
  const float* type_emb= (const float*)d_in[3];
  const float* gW1     = (const float*)d_in[4];
  const float* gb1     = (const float*)d_in[5];
  const float* gW2     = (const float*)d_in[6];
  const float* gb2     = (const float*)d_in[7];
  const float* eW0     = (const float*)d_in[8];
  const float* eb0     = (const float*)d_in[9];
  const float* eW1     = (const float*)d_in[10];
  const float* eb1     = (const float*)d_in[11];
  const float* eW2     = (const float*)d_in[12];
  const float* eb2     = (const float*)d_in[13];
  const float* nW      = (const float*)d_in[14];
  const float* nb      = (const float*)d_in[15];
  const float* cW      = (const float*)d_in[16];
  const float* cb      = (const float*)d_in[17];

  float* out = (float*)d_out;
  float* num_out   = out;                      // [1024,32]
  float* cat_out   = out + 32768;              // [1024,32,100]
  float* gates_out = out + 3309568;            // [1024,64,2]
  int*   topi_out  = (int*)(out + 3440640);    // [1024,64,2]
  int*   mask_out  = (int*)(out + 3571712);    // [1024,64]

  char* ws = (char*)d_ws;
  unsigned short* h0    = (unsigned short*)(ws + 0);          // [16,1024,1024] bf16 (32MB)
  unsigned short* h1    = (unsigned short*)(ws + 33554432);   // [16,1024,1024] bf16 (32MB)
  float*          eout  = (float*)(ws + 67108864);            // [16,1024,256]  f32 (16MB)
  float*          P1    = (float*)(ws + 83886080);            // [1024,256]
  float*          PT    = (float*)(ws + 84934656);            // [64,256]
  unsigned short* aptb  = (unsigned short*)(ws + 85000192);   // [64,1024]   bf16
  unsigned short* codb  = (unsigned short*)(ws + 85131264);   // [1024,512]  bf16
  unsigned short* eW0b  = (unsigned short*)(ws + 86179840);   // 8M elems
  unsigned short* eW1b  = (unsigned short*)(ws + 102957056);  // 16M elems
  unsigned short* eW2b  = (unsigned short*)(ws + 136511488);  // 4M elems
  unsigned short* gW1b  = (unsigned short*)(ws + 144900096);  // 1536*256
  unsigned short* cWb   = (unsigned short*)(ws + 145686528);  // 32*256*100
  unsigned short* combined = h0;  // h0 dead after h1 GEMM; reuse (32MB)

  dim3 blk(256);

  // ---- pre-pass: convert all GEMM operands to bf16 (bandwidth-bound) ----
  auto cvt = [&](const float* src, unsigned short* dst, long long n) {
    long long npairs = n / 2;
    cvt_f32_bf16_kernel<<<dim3((unsigned)((npairs + 255) / 256)), blk, 0, stream>>>(
        src, dst, npairs);
  };
  cvt(codec, codb, 1024LL * 512);
  cvt(gW1,  gW1b, 1536LL * 256);
  cvt(eW0,  eW0b, 16LL * 512 * 1024);
  cvt(eW1,  eW1b, 16LL * 1024 * 1024);
  cvt(eW2,  eW2b, 16LL * 1024 * 256);
  cvt(cW,   cWb,  32LL * 256 * 100);
  build_apt_kernel<<<256, 256, 0, stream>>>(pos_emb, type_emb, aptb);

  // ---- gate factorization GEMMs ----
  // P1 = codec @ gW1[0:512,:]           (M=1024,N=256,K=512)
  gemm_wmma_bf16<float, 0><<<dim3(2, 8, 1), blk, 0, stream>>>(
      codb, 0, 512, gW1b, 0, 256, nullptr, 0, P1, 0, 256, 1024, 256, 512);
  // PT = [pos|type] @ gW1[512:1536,:]   (M=64,N=256,K=1024)
  gemm_wmma_bf16<float, 0><<<dim3(2, 1, 1), blk, 0, stream>>>(
      aptb, 0, 1024, gW1b + 512 * 256, 0, 256, nullptr, 0, PT, 0, 256, 64, 256, 1024);

  // gate GeLU + logits + top-2 + softmax
  gate_topk_kernel<<<8192, 256, 0, stream>>>(P1, PT, gW1, gb1, gW2, gb2, maskp,
                                             gates_out, topi_out);

  // ---- expert stack, batched over E=16 via blockIdx.z ----
  // h0 = leaky(codec @ eW0[e] + eb0[e])  (1024x1024, K=512)
  gemm_wmma_bf16<unsigned short, 1><<<dim3(8, 8, 16), blk, 0, stream>>>(
      codb, 0, 512, eW0b, 512LL * 1024, 1024, eb0, 1024,
      h0, 1024LL * 1024, 1024, 1024, 1024, 512);
  // h1 = leaky(h0 @ eW1[e] + eb1[e])     (1024x1024, K=1024)
  gemm_wmma_bf16<unsigned short, 1><<<dim3(8, 8, 16), blk, 0, stream>>>(
      h0, 1024LL * 1024, 1024, eW1b, 1024LL * 1024, 1024, eb1, 1024,
      h1, 1024LL * 1024, 1024, 1024, 1024, 1024);
  // eout = h1 @ eW2[e] + eb2[e]          (1024x256, K=1024)
  gemm_wmma_bf16<float, 0><<<dim3(2, 8, 16), blk, 0, stream>>>(
      h1, 1024LL * 1024, 1024, eW2b, 1024LL * 256, 256, eb2, 256,
      eout, 1024LL * 256, 256, 1024, 256, 1024);

  // top-2 gather + weighted sum -> combined (bf16)
  combine_kernel<<<65536, 256, 0, stream>>>(eout, gates_out, topi_out, combined);

  // numerical head
  num_head_kernel<<<128, 256, 0, stream>>>(combined, nW, nb, num_out);

  // categorical head: 32 batched GEMMs [1024,256]x[256,100]
  // A rows strided by S*D across b; batch z advances the position.
  gemm_wmma_bf16<float, 0><<<dim3(1, 8, 32), blk, 0, stream>>>(
      combined + 32 * 256, 256, 64 * 256, cWb, 256LL * 100, 100, cb, 100,
      cat_out, 100, 32 * 100, 1024, 100, 256);

  // mask passthrough
  hipMemcpyAsync(mask_out, maskp, 65536 * sizeof(int),
                 hipMemcpyDeviceToDevice, stream);
}
